// TauRecursiveGNN_6176162972392
// MI455X (gfx1250) — matrix operand order repaired and checked by reading, not compiled, not run on hardware
//
#include <hip/hip_runtime.h>
#include <hip/hip_bf16.h>

// ---------------------------------------------------------------------------
// TauRecursiveGNN for MI455X (gfx1250, wave32, WMMA bf16 16x16x32)
//
//   1. pack weights (f32 -> bf16, per-lane WMMA B layout)      [once]
//   2. tau/steps per node                                      [once]
//   3. h0 = relu(x @ w_in + b_in)        via v_wmma_f32_16x16x32_bf16
//   4. 5x: zero agg; edge gather+|diff|+atomic segment sum;
//          fused GRU cell (6 WMMA accumulator chains + gates in VALU,
//          B operands block-shared through LDS)
//   5. out = h @ w_out + b_out           via WMMA
// ---------------------------------------------------------------------------

#define NNODES  50000
#define NEDGES  800000
#define IN_DIMC 128
#define HID     96
#define OUT_DIMC 64
#define MAX_REC 5
#define MTILES  3125      // NNODES / 16 (exact)
#define WPB     8         // waves per block: input/out projection
#define GWPB    4         // waves per block: GRU kernel

typedef __attribute__((ext_vector_type(16))) __bf16 v16bf;
typedef __attribute__((ext_vector_type(8)))  __bf16 v8bf;
typedef __attribute__((ext_vector_type(8)))  float  v8f;

static __device__ __forceinline__ __bf16 f2bf(float f) {
  unsigned u = __builtin_bit_cast(unsigned, f);
  u += 0x7FFFu + ((u >> 16) & 1u);               // round-to-nearest-even
  unsigned short s = (unsigned short)(u >> 16);
  return __builtin_bit_cast(__bf16, s);
}

// Build the 16-bit A-operand per-lane vector from two contiguous 8-half runs
// (ISA 16-bit A 16x32 layout: e<8 -> K = kb+e, e>=8 -> K = 16+kb+(e-8)).
static __device__ __forceinline__ v16bf combineA(const __bf16* lo, const __bf16* hi) {
  v8bf a = *(const v8bf*)lo;
  v8bf b = *(const v8bf*)hi;
  v16bf r;
#pragma unroll
  for (int i = 0; i < 8; ++i) { r[i] = a[i]; r[i + 8] = b[i]; }
  return r;
}

// ---------------------------------------------------------------------------
// Weight packing into per-lane WMMA B layout:
//   blob[((ct*nks + ks)*32 + lane)*16 + e] = B[ks*32 + (lane&16?8:0) + (e&7) + (e&8?16:0)][ct*16 + (lane&15)]
// so a lane's 16 bf16 B elements are one contiguous 32-byte load.
// ---------------------------------------------------------------------------
__global__ void pack_kn_kernel(const float* __restrict__ src, __bf16* __restrict__ dst,
                               int ncols, int nksteps, int total) {
  int tid = blockIdx.x * blockDim.x + threadIdx.x;
  if (tid >= total) return;
  int e    = tid & 15;
  int lane = (tid >> 4) & 31;
  int rest = tid >> 9;
  int ks = rest % nksteps;
  int ct = rest / nksteps;
  int n = ct * 16 + (lane & 15);
  int k = ks * 32 + ((lane & 16) ? 8 : 0) + (e & 7) + ((e & 8) ? 16 : 0);
  dst[tid] = f2bf(src[k * ncols + n]);          // src row-major [K][ncols]
}

__global__ void pack_nk_kernel(const float* __restrict__ src, __bf16* __restrict__ dst,
                               int kdim, int nksteps, int total) {
  // src row-major [Ntotal][kdim]; B[k][n] = src[n][k]  (i.e. W^T)
  int tid = blockIdx.x * blockDim.x + threadIdx.x;
  if (tid >= total) return;
  int e    = tid & 15;
  int lane = (tid >> 4) & 31;
  int rest = tid >> 9;
  int ks = rest % nksteps;
  int ct = rest / nksteps;
  int n = ct * 16 + (lane & 15);
  int k = ks * 32 + ((lane & 16) ? 8 : 0) + (e & 7) + ((e & 8) ? 16 : 0);
  dst[tid] = f2bf(src[n * kdim + k]);
}

// ---------------------------------------------------------------------------
// tau / steps:  steps = min(int(1 / softplus(relu(x@w1+b1)@w2+b2)), 5)
// ---------------------------------------------------------------------------
__global__ void tau_steps_kernel(const float* __restrict__ x,
                                 const float* __restrict__ w1, const float* __restrict__ b1,
                                 const float* __restrict__ w2, const float* __restrict__ b2,
                                 int* __restrict__ steps) {
  __shared__ float sw1[IN_DIMC * 16];
  __shared__ float sw2[16];
  for (int t = threadIdx.x; t < IN_DIMC * 16; t += blockDim.x) sw1[t] = w1[t];
  if (threadIdx.x < 16) sw2[threadIdx.x] = w2[threadIdx.x];
  __syncthreads();
  int nid = blockIdx.x * blockDim.x + threadIdx.x;
  if (nid >= NNODES) return;
  float acc[16];
#pragma unroll
  for (int j = 0; j < 16; ++j) acc[j] = b1[j];
  const float* xr = x + (size_t)nid * IN_DIMC;
  for (int k = 0; k < IN_DIMC; ++k) {
    float xv = xr[k];
#pragma unroll
    for (int j = 0; j < 16; ++j) acc[j] = fmaf(xv, sw1[k * 16 + j], acc[j]);
  }
  float s = b2[0];
#pragma unroll
  for (int j = 0; j < 16; ++j) s = fmaf(fmaxf(acc[j], 0.f), sw2[j], s);
  float sp  = (s > 20.f) ? s : log1pf(__expf(s));   // softplus, overflow-safe
  float inv = 1.f / sp;
  int st = (inv >= (float)MAX_REC) ? MAX_REC : (int)inv;   // trunc toward zero
  steps[nid] = st;
}

// ---------------------------------------------------------------------------
// h0 = relu(x @ w_in + b_in)   — one wave per 16-node tile, K=128 (4 steps)
// B blob (24KB) staged once into LDS and shared by all 8 waves.
// ---------------------------------------------------------------------------
__global__ __launch_bounds__(32 * WPB)
void input_proj_kernel(const float* __restrict__ x,
                       const __bf16* __restrict__ packB,   // [6][4][32][16]
                       const float* __restrict__ bias,
                       float* __restrict__ h) {
  __shared__ __align__(16) __bf16 ldsX[WPB][16][IN_DIMC];  // 32 KB
  __shared__ __align__(32) __bf16 ldsB[6 * 4 * 512];       // 24 KB
  for (int t = threadIdx.x; t < 1536; t += 32 * WPB)
    ((uint4*)ldsB)[t] = ((const uint4*)packB)[t];
  __syncthreads();

  int lane  = threadIdx.x & 31;
  int w     = threadIdx.x >> 5;
  int mtile = blockIdx.x * WPB + w;
  if (mtile >= MTILES) return;                 // no barriers past this point
  int m0 = mtile * 16;
  for (int t = lane; t < 16 * IN_DIMC; t += 32) {
    int r = t >> 7, c = t & 127;
    ldsX[w][r][c] = f2bf(x[(size_t)(m0 + r) * IN_DIMC + c]);
  }
  int mrow = lane & 15;
  int kb   = (lane & 16) ? 8 : 0;
  v16bf a[4];
#pragma unroll
  for (int ks = 0; ks < 4; ++ks)
    a[ks] = combineA(&ldsX[w][mrow][ks * 32 + kb], &ldsX[w][mrow][ks * 32 + 16 + kb]);
  int msel = (lane & 16) ? 8 : 0;
#pragma unroll
  for (int ct = 0; ct < 6; ++ct) {
    int nIdx = ct * 16 + (lane & 15);
    float bv = bias[nIdx];
    v8f acc;
#pragma unroll
    for (int i = 0; i < 8; ++i) acc[i] = bv;
#pragma unroll
    for (int ks = 0; ks < 4; ++ks) {
      v16bf b = *(const v16bf*)&ldsB[(((ct * 4 + ks) * 32 + lane) << 4)];
      acc = __builtin_amdgcn_wmma_f32_16x16x32_bf16(false, a[ks], false, b,
                                                    (short)0, acc, false, false);
    }
#pragma unroll
    for (int i = 0; i < 8; ++i) {
      int node = m0 + i + msel;
      h[(size_t)node * HID + nIdx] = fmaxf(acc[i], 0.f);
    }
  }
}

// ---------------------------------------------------------------------------
__global__ void zero_kernel(float* __restrict__ p, int n) {
  int stride = gridDim.x * blockDim.x;
  for (int i = blockIdx.x * blockDim.x + threadIdx.x; i < n; i += stride) p[i] = 0.f;
}

// ---------------------------------------------------------------------------
// Edge phase: one wave32 per edge, 3 column steps of 32 lanes.
// agg[row] += |h[row] - h[col]| via global_atomic_add_f32 (L2-resident h).
// Edges whose destination node is frozen (steps<=iter) are skipped.
// ---------------------------------------------------------------------------
__global__ void edge_kernel(const float* __restrict__ h,
                            const int* __restrict__ eidx,
                            const int* __restrict__ steps,
                            int iter,
                            float* __restrict__ agg) {
  int gw = (blockIdx.x * blockDim.x + threadIdx.x) >> 5;
  if (gw >= NEDGES) return;
  int lane = threadIdx.x & 31;
  int r = eidx[gw];
  if (steps[r] <= iter) return;                 // uniform per wave
  int c = eidx[NEDGES + gw];
  const float* hr = h + (size_t)r * HID;
  const float* hc = h + (size_t)c * HID;
  float* ar = agg + (size_t)r * HID;
  if (lane == 0) { __builtin_prefetch(hc, 0, 0); }   // global_prefetch_b8
#pragma unroll
  for (int s = 0; s < 3; ++s) {
    int col = s * 32 + lane;
    float d = fabsf(hr[col] - hc[col]);
    atomicAdd(&ar[col], d);
  }
}

// ---------------------------------------------------------------------------
// Fused GRU cell: gi = agg@wih^T + bih, gh = h@whh^T + bhh, gates, mask.
// One wave per 16-node tile; per 16-col slice the 18 KB of B tiles
// (r/z/n x {input,hidden} x 3 K-steps) are cooperatively staged into LDS and
// shared by all 4 waves -> 18 WMMAs per slice feed from ds_load_b128.
// ---------------------------------------------------------------------------
__global__ __launch_bounds__(32 * GWPB)
void gru_kernel(const float* __restrict__ agg,
                const float* __restrict__ hcur,
                const __bf16* __restrict__ packI,   // [18][3][32][16]
                const __bf16* __restrict__ packH,
                const float* __restrict__ bih,
                const float* __restrict__ bhh,
                const int* __restrict__ steps,
                int iter,
                float* __restrict__ hnext) {
  __shared__ __align__(16) __bf16 ldsAgg[GWPB][16][HID];   // 12 KB
  __shared__ __align__(16) __bf16 ldsH[GWPB][16][HID];     // 12 KB
  __shared__ __align__(32) __bf16 ldsBI[9][512];           //  9 KB (g*3+ks)
  __shared__ __align__(32) __bf16 ldsBH[9][512];           //  9 KB

  int lane  = threadIdx.x & 31;
  int w     = threadIdx.x >> 5;
  int mtile = blockIdx.x * GWPB + w;
  bool active = (mtile < MTILES);               // wave-uniform
  int m0 = mtile * 16;

  int mrow = lane & 15;
  int kb   = (lane & 16) ? 8 : 0;
  int msel = (lane & 16) ? 8 : 0;
  v16bf aA[3] = {}, aH[3] = {};
  int stp[8] = {};

  if (active) {
    for (int t = lane; t < 16 * HID; t += 32) {
      int r = t / HID, c = t % HID;
      size_t gidx = (size_t)(m0 + r) * HID + c;
      ldsAgg[w][r][c] = f2bf(agg[gidx]);
      ldsH[w][r][c]   = f2bf(hcur[gidx]);
    }
#pragma unroll
    for (int ks = 0; ks < 3; ++ks) {
      aA[ks] = combineA(&ldsAgg[w][mrow][ks * 32 + kb], &ldsAgg[w][mrow][ks * 32 + 16 + kb]);
      aH[ks] = combineA(&ldsH[w][mrow][ks * 32 + kb],   &ldsH[w][mrow][ks * 32 + 16 + kb]);
    }
#pragma unroll
    for (int i = 0; i < 8; ++i) stp[i] = steps[m0 + i + msel];
  }

  for (int ct = 0; ct < 6; ++ct) {
    // ---- cooperative B staging for this column slice (all threads) ----
    for (int t = threadIdx.x; t < 1152; t += 32 * GWPB) {   // 18 chunks x 64 uint4
      int chunk = t >> 6;           // 0..17
      int v     = t & 63;
      int mtx   = (chunk >= 9);     // 0 = input-path, 1 = hidden-path
      int gk    = chunk % 9;        // g*3 + ks
      int g = gk / 3, ks = gk % 3;
      const uint4* src = (const uint4*)((mtx ? packH : packI) +
                          ((size_t)(((g * 6 + ct) * 3 + ks)) << 9));
      uint4* dst = (uint4*)((mtx ? ldsBH : ldsBI)[gk]);
      dst[v] = src[v];
    }
    __syncthreads();

    if (active) {
      int nIdx = ct * 16 + (lane & 15);
      v8f accI[3], accH[3];
#pragma unroll
      for (int g = 0; g < 3; ++g) {
        float bi = bih[g * HID + nIdx];
        float bh = bhh[g * HID + nIdx];
#pragma unroll
        for (int i = 0; i < 8; ++i) { accI[g][i] = bi; accH[g][i] = bh; }
      }
#pragma unroll
      for (int g = 0; g < 3; ++g) {
#pragma unroll
        for (int ks = 0; ks < 3; ++ks) {
          v16bf bI = *(const v16bf*)&ldsBI[g * 3 + ks][lane << 4];
          v16bf bH = *(const v16bf*)&ldsBH[g * 3 + ks][lane << 4];
          accI[g] = __builtin_amdgcn_wmma_f32_16x16x32_bf16(false, aA[ks], false, bI,
                                                            (short)0, accI[g], false, false);
          accH[g] = __builtin_amdgcn_wmma_f32_16x16x32_bf16(false, aH[ks], false, bH,
                                                            (short)0, accH[g], false, false);
        }
      }
#pragma unroll
      for (int i = 0; i < 8; ++i) {
        int node = m0 + i + msel;
        float ir = accI[0][i], iz = accI[1][i], inn = accI[2][i];
        float hr = accH[0][i], hz = accH[1][i], hn  = accH[2][i];
        float rg = 1.f / (1.f + __expf(-(ir + hr)));
        float zg = 1.f / (1.f + __expf(-(iz + hz)));
        float t  = inn + rg * hn;
        float ng = 2.f / (1.f + __expf(-2.f * t)) - 1.f;   // tanh
        float hold = hcur[(size_t)node * HID + nIdx];
        float hnew = (1.f - zg) * ng + zg * hold;
        hnext[(size_t)node * HID + nIdx] = (stp[i] > iter) ? hnew : hold;
      }
    }
    __syncthreads();   // protect B buffers before next slice restages them
  }
}

// ---------------------------------------------------------------------------
// out = h @ w_out + b_out     K=96 (3 steps), 4 column tiles of 16
// B blob (12KB) staged once into LDS.
// ---------------------------------------------------------------------------
__global__ __launch_bounds__(32 * WPB)
void out_proj_kernel(const float* __restrict__ h,
                     const __bf16* __restrict__ packB,   // [4][3][32][16]
                     const float* __restrict__ bias,
                     float* __restrict__ out) {
  __shared__ __align__(16) __bf16 ldsH[WPB][16][HID];    // 24 KB
  __shared__ __align__(32) __bf16 ldsB[4 * 3 * 512];     // 12 KB
  for (int t = threadIdx.x; t < 768; t += 32 * WPB)
    ((uint4*)ldsB)[t] = ((const uint4*)packB)[t];
  __syncthreads();

  int lane  = threadIdx.x & 31;
  int w     = threadIdx.x >> 5;
  int mtile = blockIdx.x * WPB + w;
  if (mtile >= MTILES) return;
  int m0 = mtile * 16;
  for (int t = lane; t < 16 * HID; t += 32) {
    int r = t / HID, c = t % HID;
    ldsH[w][r][c] = f2bf(h[(size_t)(m0 + r) * HID + c]);
  }
  int mrow = lane & 15;
  int kb   = (lane & 16) ? 8 : 0;
  v16bf a[3];
#pragma unroll
  for (int ks = 0; ks < 3; ++ks)
    a[ks] = combineA(&ldsH[w][mrow][ks * 32 + kb], &ldsH[w][mrow][ks * 32 + 16 + kb]);
  int msel = (lane & 16) ? 8 : 0;
#pragma unroll
  for (int ct = 0; ct < 4; ++ct) {
    int nIdx = ct * 16 + (lane & 15);
    float bv = bias[nIdx];
    v8f acc;
#pragma unroll
    for (int i = 0; i < 8; ++i) acc[i] = bv;
#pragma unroll
    for (int ks = 0; ks < 3; ++ks) {
      v16bf b = *(const v16bf*)&ldsB[(((ct * 3 + ks) * 32 + lane) << 4)];
      acc = __builtin_amdgcn_wmma_f32_16x16x32_bf16(false, a[ks], false, b,
                                                    (short)0, acc, false, false);
    }
#pragma unroll
    for (int i = 0; i < 8; ++i) {
      int node = m0 + i + msel;
      out[(size_t)node * OUT_DIMC + nIdx] = acc[i];
    }
  }
}

// ---------------------------------------------------------------------------
extern "C" void kernel_launch(void* const* d_in, const int* in_sizes, int n_in,
                              void* d_out, int out_size, void* d_ws, size_t ws_size,
                              hipStream_t stream) {
  (void)in_sizes; (void)n_in; (void)out_size; (void)ws_size;
  const float* x       = (const float*)d_in[0];
  const int*   eidx    = (const int*)  d_in[1];
  const float* w_in    = (const float*)d_in[2];
  const float* b_in    = (const float*)d_in[3];
  const float* tau_w1  = (const float*)d_in[4];
  const float* tau_b1  = (const float*)d_in[5];
  const float* tau_w2  = (const float*)d_in[6];
  const float* tau_b2  = (const float*)d_in[7];
  const float* gru_wih = (const float*)d_in[8];
  const float* gru_whh = (const float*)d_in[9];
  const float* gru_bih = (const float*)d_in[10];
  const float* gru_bhh = (const float*)d_in[11];
  const float* w_out   = (const float*)d_in[12];
  const float* b_out   = (const float*)d_in[13];
  float* out = (float*)d_out;

  // workspace carve-out (~58 MB): h0, h1, agg, steps, packed bf16 weights
  char* p = (char*)d_ws;
  auto carve = [&](size_t bytes) -> char* {
    char* r = p; p += (bytes + 255) & ~(size_t)255; return r;
  };
  float*  h0    = (float*) carve((size_t)NNODES * HID * 4);
  float*  h1    = (float*) carve((size_t)NNODES * HID * 4);
  float*  agg   = (float*) carve((size_t)NNODES * HID * 4);
  int*    steps = (int*)   carve((size_t)NNODES * 4);
  __bf16* pWin  = (__bf16*)carve((size_t)6  * 4 * 512 * 2);
  __bf16* pWih  = (__bf16*)carve((size_t)18 * 3 * 512 * 2);
  __bf16* pWhh  = (__bf16*)carve((size_t)18 * 3 * 512 * 2);
  __bf16* pWout = (__bf16*)carve((size_t)4  * 3 * 512 * 2);

  // 1) pack weights into WMMA B-operand layout
  pack_kn_kernel<<<(12288 + 255) / 256, 256, 0, stream>>>(w_in,    pWin,  HID,      4, 12288);
  pack_nk_kernel<<<(27648 + 255) / 256, 256, 0, stream>>>(gru_wih, pWih,  HID,      3, 27648);
  pack_nk_kernel<<<(27648 + 255) / 256, 256, 0, stream>>>(gru_whh, pWhh,  HID,      3, 27648);
  pack_kn_kernel<<<(6144  + 255) / 256, 256, 0, stream>>>(w_out,   pWout, OUT_DIMC, 3, 6144);

  // 2) per-node recursion depth
  tau_steps_kernel<<<(NNODES + 255) / 256, 256, 0, stream>>>(
      x, tau_w1, tau_b1, tau_w2, tau_b2, steps);

  // 3) input projection (WMMA)
  const int projBlocks = (MTILES + WPB - 1) / WPB;     // 391
  input_proj_kernel<<<projBlocks, 32 * WPB, 0, stream>>>(x, pWin, b_in, h0);

  // 4) recursion
  const int gruBlocks = (MTILES + GWPB - 1) / GWPB;    // 782
  float* hc = h0;
  float* hn = h1;
  for (int it = 0; it < MAX_REC; ++it) {
    zero_kernel<<<2048, 256, 0, stream>>>(agg, NNODES * HID);
    edge_kernel<<<NEDGES / 8, 256, 0, stream>>>(hc, eidx, steps, it, agg);
    gru_kernel<<<gruBlocks, 32 * GWPB, 0, stream>>>(agg, hc, pWih, pWhh,
                                                    gru_bih, gru_bhh, steps, it, hn);
    float* tmp = hc; hc = hn; hn = tmp;
  }

  // 5) output projection (WMMA)
  out_proj_kernel<<<projBlocks, 32 * WPB, 0, stream>>>(hc, pWout, b_out, out);
}
